// EncoderGRUODE_12618613915681
// MI455X (gfx1250) — matrix operand order repaired and checked by reading, not compiled, not run on hardware
//
#include <hip/hip_runtime.h>

typedef __attribute__((ext_vector_type(16))) _Float16 v16h;
typedef __attribute__((ext_vector_type(8)))  float    v8f;

namespace {
constexpr int T_ = 1000;
constexpr int B_ = 256;
constexpr int D_ = 64;
constexpr int H_ = 128;          // hidden = 2*LATENT
constexpr int MT = 16;           // batch rows per workgroup (one WMMA M-tile)
constexpr int SH = H_ + 8;       // padded LDS row stride (halves) for H-wide tiles
constexpr int SX = D_ + 8;       // padded LDS row stride (halves) for x tile
constexpr int KCH = H_ / 32;     // K chunks for K=128 (f16 wmma k=32)
constexpr int KCD = D_ / 32;     // K chunks for K=64

// Workspace layout (halves), all weights pre-swizzled into B-fragment order
constexpr int OFF_WIH  = 0;                   // 384 x 64
constexpr int OFF_WHH  = OFF_WIH + 384 * 64;  // 384 x 128
constexpr int OFF_W1   = OFF_WHH + 384 * 128; // 128 x 128
constexpr int OFF_W2   = OFF_W1  + 128 * 128; // 128 x 128
constexpr int OFF_WOUT = OFF_W2  + 128 * 128; // 128 x 128
constexpr int SWZ_HALVES = OFF_WOUT + 128 * 128;       // 122880 halves
constexpr int WEIGHT_LDS_HALVES = OFF_WOUT;            // 106496 halves in LDS
constexpr size_t DTS_BYTE_OFF = (size_t)SWZ_HALVES * 2;  // fp32 dts after halves

constexpr size_t LDS_BYTES =
    (size_t)(WEIGHT_LDS_HALVES + 3 * MT * SH + MT * SX) * 2  // halves
    + 3 * MT * 4;                                            // mf, rf, seen
}

#define WMMA_F16(ACC, A, Bf)                                                   \
  ACC = __builtin_amdgcn_wmma_f32_16x16x32_f16(false, (A), false, (Bf),        \
                                               (short)0, (ACC), false, false)

__device__ __forceinline__ v8f zero8() {
  v8f z;
#pragma unroll
  for (int i = 0; i < 8; ++i) z[i] = 0.f;
  return z;
}

// Branch-free fast nonlinearities (keep EXEC uniform on the serial path):
// v_exp_f32 + v_rcp_f32, saturates correctly via exp over/underflow.
__device__ __forceinline__ float fast_sigmoid(float x) {
  return __builtin_amdgcn_rcpf(1.f + __expf(-x));
}
__device__ __forceinline__ float fast_tanh(float x) {
  return 2.f * __builtin_amdgcn_rcpf(1.f + __expf(-2.f * x)) - 1.f;
}

// Gather a 16x32 f16 A-fragment from a row-major LDS tile.
// ISA layout: lane = M row (lanes 0-15 / 16-31 split K halves);
// element e -> k = (e&7) + (e>=8 ? 16 : 0) + 8*(lane>=16)
// e=0..7 and e=8..15 are LDS-contiguous -> compiler emits 2x ds_load_b128.
__device__ __forceinline__ v16h load_a_frag(const _Float16* __restrict__ m,
                                            int stride, int c, int lane) {
  const int row = lane & 15;
  const int hlf = lane >> 4;
  const _Float16* base = m + row * stride + 32 * c + 8 * hlf;
  v16h a;
#pragma unroll
  for (int e = 0; e < 16; ++e) {
    const int k = (e & 7) + ((e & 8) << 1);
    a[e] = base[k];
  }
  return a;
}

// Pre-swizzled B fragment: 32 lanes x 16 contiguous halves per fragment
__device__ __forceinline__ v16h load_b_frag(const _Float16* __restrict__ p,
                                            int frag, int lane) {
  return *reinterpret_cast<const v16h*>(p + (size_t)(frag * 32 + lane) * 16);
}

// ---- prologue: fp32 W[N][K] -> f16 B-fragment-swizzled buffer -------------
// dst[((tile*KC + c)*32 + lane)*16 + e] = W[tile*16 + lane%16][32c + 8*(lane/16) + (e&7) + 16*(e>=8)]
__global__ void swizzle_w_kernel(const float* __restrict__ W,
                                 _Float16* __restrict__ dst, int N, int K) {
  const int idx = blockIdx.x * blockDim.x + threadIdx.x;
  if (idx >= N * K) return;
  const int e    = idx & 15;
  const int lane = (idx >> 4) & 31;
  const int frag = idx >> 9;
  const int KC   = K >> 5;
  const int c    = frag % KC;
  const int tile = frag / KC;
  const int row  = tile * 16 + (lane & 15);
  const int col  = 32 * c + ((lane >> 4) << 3) + (e & 7) + ((e & 8) << 1);
  dst[idx] = (_Float16)W[row * K + col];
}

__global__ void make_dts_kernel(const float* __restrict__ tps,
                                float* __restrict__ dts, int T) {
  const int t = blockIdx.x * blockDim.x + threadIdx.x;
  if (t < T) dts[t] = (t == 0) ? 0.0f : (tps[t] - tps[t - 1]);
}

// ---- main persistent scan kernel ------------------------------------------
extern __shared__ char smem_raw[];

__global__ void __launch_bounds__(256)
gruode_scan_kernel(const float* __restrict__ x, const int* __restrict__ mask,
                   const float* __restrict__ dts,
                   const _Float16* __restrict__ swz,
                   const float* __restrict__ b_ih, const float* __restrict__ b_hh,
                   const float* __restrict__ node_b1,
                   const float* __restrict__ node_b2,
                   const float* __restrict__ b_out, float* __restrict__ out) {
  _Float16* lds  = reinterpret_cast<_Float16*>(smem_raw);
  _Float16* sWih = lds + OFF_WIH;
  _Float16* sWhh = lds + OFF_WHH;
  _Float16* sW1  = lds + OFF_W1;
  _Float16* sW2  = lds + OFF_W2;
  _Float16* h16  = lds + WEIGHT_LDS_HALVES;  // 16 x SH, current hidden state
  _Float16* g16  = h16 + MT * SH;            // tanh(h W1^T + b1)
  _Float16* ho16 = g16 + MT * SH;            // h_ode
  _Float16* x16  = ho16 + MT * SH;           // 16 x SX, x_t tile
  float* mf  = reinterpret_cast<float*>(x16 + MT * SX);
  float* rf  = mf + MT;
  int* seen  = reinterpret_cast<int*>(rf + MT);

  const int tid  = threadIdx.x;
  const int lane = tid & 31;
  const int wave = tid >> 5;     // 8 waves; wave w owns output columns [16w,16w+16)
  const int b0   = blockIdx.x * MT;
  const int nloc = lane & 15;
  const int hlf  = lane >> 4;
  const int ncol = wave * 16 + nloc;

  // Bulk-copy all swizzled weights into LDS (212992 B, uint4 vectorized)
  {
    const uint4* src = reinterpret_cast<const uint4*>(swz);
    uint4* dst = reinterpret_cast<uint4*>(lds);
    for (int i = tid; i < (WEIGHT_LDS_HALVES * 2) / 16; i += 256) dst[i] = src[i];
  }
  for (int i = tid; i < MT * SH; i += 256) h16[i] = (_Float16)0.f;  // h0 = 0
  if (tid < MT) seen[tid] = 0;

  // Loop-invariant per-lane biases (depend only on output column)
  const float b1v  = node_b1[ncol];
  const float b2v  = node_b2[ncol];
  const float brz  = b_ih[ncol] + b_hh[ncol];              // r-gate combined
  const float bzz  = b_ih[H_ + ncol] + b_hh[H_ + ncol];    // z-gate combined
  const float binv = b_ih[2 * H_ + ncol];                  // i_n
  const float bhnv = b_hh[2 * H_ + ncol];                  // h_n
  const float bov  = b_out[ncol];

  v8f h_reg = zero8();   // wave-owned column block of h
  v8f hlast = zero8();   // h at last observed step

  __syncthreads();

  for (int t = 0; t < T_; ++t) {
    // ---- stage mask / right-fill / x_t tile (+ prefetch next x_t) ----
    if (tid < MT) {
      const int m = mask[(size_t)(b0 + tid) * T_ + t];
      rf[tid] = seen[tid] ? 1.f : 0.f;  // observed strictly before t
      mf[tid] = (float)m;
      seen[tid] |= m;
      if (t + 1 < T_)
        __builtin_prefetch(x + ((size_t)(b0 + tid) * T_ + (t + 1)) * D_, 0, 0);
    }
    {
      const float* xb = x + (size_t)b0 * T_ * D_ + (size_t)t * D_;
#pragma unroll
      for (int i = 0; i < (MT * D_) / 256; ++i) {
        const int idx = tid + 256 * i;
        const int r = idx >> 6, cc = idx & 63;
        x16[r * SX + cc] = (_Float16)xb[(size_t)r * T_ * D_ + cc];
      }
    }
    const float dt = dts[t];
    __syncthreads();

    // ---- phase A: g = tanh(h @ W1^T + b1) ----
    v8f acc = zero8();
#pragma unroll
    for (int c = 0; c < KCH; ++c) {
      v16h a = load_a_frag(h16, SH, c, lane);
      WMMA_F16(acc, a, load_b_frag(sW1, wave * KCH + c, lane));
    }
#pragma unroll
    for (int j = 0; j < 8; ++j)
      g16[(j + 8 * hlf) * SH + ncol] = (_Float16)fast_tanh(acc[j] + b1v);
    __syncthreads();

    // ---- phase B: f = g @ W2^T + b2 ; h_ode = freeze(h + dt*f) ----
    acc = zero8();
#pragma unroll
    for (int c = 0; c < KCH; ++c) {
      v16h a = load_a_frag(g16, SH, c, lane);
      WMMA_F16(acc, a, load_b_frag(sW2, wave * KCH + c, lane));
    }
    v8f ho;
#pragma unroll
    for (int j = 0; j < 8; ++j) {
      const float rr = rf[j + 8 * hlf];
      const float hv = h_reg[j];
      float hov = hv + dt * (acc[j] + b2v);
      hov = rr * hov + (1.f - rr) * hv;  // frozen before first observation
      ho[j] = hov;
      ho16[(j + 8 * hlf) * SH + ncol] = (_Float16)hov;
    }
    __syncthreads();

    // ---- phase C: GRU gates. r/z accumulate input+hidden into one C frag ----
    v8f ar = zero8(), az = zero8(), ain = zero8(), ahn = zero8();
#pragma unroll
    for (int c = 0; c < KCH; ++c) {
      v16h a = load_a_frag(ho16, SH, c, lane);
      WMMA_F16(ar,  a, load_b_frag(sWhh, (0 * 8 + wave) * KCH + c, lane));
      WMMA_F16(az,  a, load_b_frag(sWhh, (1 * 8 + wave) * KCH + c, lane));
      WMMA_F16(ahn, a, load_b_frag(sWhh, (2 * 8 + wave) * KCH + c, lane));
    }
#pragma unroll
    for (int c = 0; c < KCD; ++c) {
      v16h a = load_a_frag(x16, SX, c, lane);
      WMMA_F16(ar,  a, load_b_frag(sWih, (0 * 8 + wave) * KCD + c, lane));
      WMMA_F16(az,  a, load_b_frag(sWih, (1 * 8 + wave) * KCD + c, lane));
      WMMA_F16(ain, a, load_b_frag(sWih, (2 * 8 + wave) * KCD + c, lane));
    }
#pragma unroll
    for (int j = 0; j < 8; ++j) {
      const int row = j + 8 * hlf;
      const float r = fast_sigmoid(ar[j] + brz);
      const float z = fast_sigmoid(az[j] + bzz);
      const float n = fast_tanh(ain[j] + binv + r * (ahn[j] + bhnv));
      const float hr = (1.f - z) * n + z * ho[j];
      const float m  = mf[row];
      const float hv = m * hr + (1.f - m) * ho[j];
      h_reg[j] = hv;
      hlast[j] = m * hv + (1.f - m) * hlast[j];  // track last observed state
      h16[row * SH + ncol] = (_Float16)hv;
    }
    __syncthreads();
  }

  // ---- epilogue: out = h_last @ W_out^T + b_out ----
#pragma unroll
  for (int j = 0; j < 8; ++j)
    h16[(j + 8 * hlf) * SH + ncol] = (_Float16)hlast[j];
  __syncthreads();
  const _Float16* swzWout = swz + OFF_WOUT;  // read directly from global (once)
  v8f acco = zero8();
#pragma unroll
  for (int c = 0; c < KCH; ++c) {
    v16h a = load_a_frag(h16, SH, c, lane);
    v16h b = *reinterpret_cast<const v16h*>(
        swzWout + (size_t)((wave * KCH + c) * 32 + lane) * 16);
    WMMA_F16(acco, a, b);
  }
#pragma unroll
  for (int j = 0; j < 8; ++j)
    out[(size_t)(b0 + j + 8 * hlf) * H_ + ncol] = acco[j] + bov;
}

extern "C" void kernel_launch(void* const* d_in, const int* in_sizes, int n_in,
                              void* d_out, int out_size, void* d_ws,
                              size_t ws_size, hipStream_t stream) {
  (void)in_sizes; (void)n_in; (void)out_size; (void)ws_size;
  const float* x     = (const float*)d_in[0];
  const float* tps   = (const float*)d_in[1];
  const int*   mask  = (const int*)d_in[2];
  const float* W_ih  = (const float*)d_in[3];
  const float* W_hh  = (const float*)d_in[4];
  const float* b_ih  = (const float*)d_in[5];
  const float* b_hh  = (const float*)d_in[6];
  const float* W1    = (const float*)d_in[7];
  const float* nb1   = (const float*)d_in[8];
  const float* W2    = (const float*)d_in[9];
  const float* nb2   = (const float*)d_in[10];
  const float* W_out = (const float*)d_in[11];
  const float* b_out = (const float*)d_in[12];

  _Float16* swz = (_Float16*)d_ws;
  float* dts = (float*)((char*)d_ws + DTS_BYTE_OFF);

  // Prologue: pre-swizzle weights to f16 B-fragment layout + build dt vector
  swizzle_w_kernel<<<(384 * 64 + 255) / 256, 256, 0, stream>>>(W_ih, swz + OFF_WIH, 384, 64);
  swizzle_w_kernel<<<(384 * 128 + 255) / 256, 256, 0, stream>>>(W_hh, swz + OFF_WHH, 384, 128);
  swizzle_w_kernel<<<(128 * 128 + 255) / 256, 256, 0, stream>>>(W1, swz + OFF_W1, 128, 128);
  swizzle_w_kernel<<<(128 * 128 + 255) / 256, 256, 0, stream>>>(W2, swz + OFF_W2, 128, 128);
  swizzle_w_kernel<<<(128 * 128 + 255) / 256, 256, 0, stream>>>(W_out, swz + OFF_WOUT, 128, 128);
  make_dts_kernel<<<(T_ + 255) / 256, 256, 0, stream>>>(tps, dts, T_);

  // Persistent scan: 16 workgroups (16 batch rows each), 8 waves, ~227 KB LDS
  gruode_scan_kernel<<<B_ / MT, 256, LDS_BYTES, stream>>>(
      x, mask, dts, swz, b_ih, b_hh, nb1, nb2, b_out, (float*)d_out);
}
